// VPEncoderLayerByDeForAtt_84061099917454
// MI455X (gfx1250) — matrix operand order repaired and checked by reading, not compiled, not run on hardware
//
#include <hip/hip_runtime.h>
#include <hip/hip_bf16.h>
#include <math.h>

#define EMBED  256
#define HEADS  8
#define LEVELS 4
#define POINTS 4
#define HD     32
#define NVQ    13294
#define BSZ    2
#define ROWS   (BSZ * NVQ)   // 26588

typedef __attribute__((ext_vector_type(16))) _Float16 v16h;
typedef __attribute__((ext_vector_type(8)))  _Float16 v8h;
typedef __attribute__((ext_vector_type(8)))  float    v8f;

// ---------------- prep: q = query + query_pos -> f16 ; value -> f16 ----------------
__global__ void prep_kernel(const float* __restrict__ query,
                            const float* __restrict__ qpos,
                            const float* __restrict__ value,
                            _Float16* __restrict__ q16,
                            _Float16* __restrict__ v16,
                            long n) {
  long i = (long)blockIdx.x * blockDim.x + threadIdx.x;
  if (i < n) {
    q16[i] = (_Float16)(query[i] + qpos[i]);
    v16[i] = (_Float16)value[i];
  }
}

// ---------------- weight reshuffle into WMMA B-fragment layout ----------------
// frag[((kt*NT + nt)*32 + lane)*16 + j] = W[kt*32 + (lane>=16?16:0) + j][nt*16 + (lane&15)]
__global__ void wprep_kernel(const float* __restrict__ W,
                             _Float16* __restrict__ frag, int N) {
  int i = blockIdx.x * 256 + threadIdx.x;
  if (i >= 256 * N) return;
  int j    = i & 15;
  int lane = (i >> 4) & 31;
  int rest = i >> 9;
  int NT = N >> 4;
  int nt = rest % NT;
  int kt = rest / NT;
  int k = kt * 32 + (lane >> 4) * 16 + j;
  int n = nt * 16 + (lane & 15);
  frag[i] = (_Float16)W[k * N + n];
}

// ---------------- generic WMMA GEMM: C = A(Mx256,f16) x W(256xN) + bias ----------------
enum { EP_VALUE = 0, EP_OFF = 1, EP_ATTN = 2, EP_OUT = 3, EP_FFN = 4 };

template<int MODE>
__global__ __launch_bounds__(128)
void gemm_wmma_kernel(const _Float16* __restrict__ A,
                      const _Float16* __restrict__ Bfrag,
                      const float* __restrict__ bias,
                      float* __restrict__ outF32,
                      _Float16* __restrict__ outF16,
                      const float* __restrict__ resid,
                      const unsigned char* __restrict__ mask,
                      int M, int N) {
  // 8 ksteps x 4 ntiles x (32 lanes * 16 halves) = 32 KB of weight fragments,
  // identical for all 4 waves of the block -> stage once in LDS.
  __shared__ _Float16 Bs[8 * 4 * 512];
  const int NT  = N >> 4;
  const int n64 = blockIdx.y;
  for (int kt = 0; kt < 8; ++kt) {
    const v8h* src = (const v8h*)(Bfrag + (size_t)(kt * NT + n64 * 4) * 512);
    v8h* dst = (v8h*)(Bs + kt * 2048);
    for (int i = threadIdx.x; i < 256; i += 128) dst[i] = src[i];
  }
  __syncthreads();

  const int lane = threadIdx.x & 31;
  const int wid  = threadIdx.x >> 5;
  const int mtile  = blockIdx.x * 4 + wid;
  const int Mtiles = (M + 15) >> 4;
  if (mtile >= Mtiles) return;   // whole-wave uniform: EXEC stays all-ones for WMMA
  const int half = lane >> 4;
  const int l16  = lane & 15;

  int rowA = mtile * 16 + l16;
  if (rowA >= M) rowA = M - 1;   // clamp loads; stores guarded below
  const _Float16* arow = A + (size_t)rowA * 256;

  const v8f vzero = {0.f, 0.f, 0.f, 0.f, 0.f, 0.f, 0.f, 0.f};
  v8f acc[4] = {vzero, vzero, vzero, vzero};

  for (int kt = 0; kt < 8; ++kt) {
    const int k0 = kt * 32;
    // A fragment, contiguous K order matching the B fragments:
    //   lane<16:  M=lane,    halves 0..15 = K k0..k0+15
    //   lane>=16: M=lane-16, halves 0..15 = K k0+16..k0+31
    v8h alo = *(const v8h*)(arow + k0 + half * 16);
    v8h ahi = *(const v8h*)(arow + k0 + half * 16 + 8);
    v16h a;
#pragma unroll
    for (int i = 0; i < 8; ++i) { a[i] = alo[i]; a[i + 8] = ahi[i]; }
    if (kt < 7) __builtin_prefetch(arow + k0 + 32, 0, 1);
#pragma unroll
    for (int t = 0; t < 4; ++t) {
      const v8h* bp = (const v8h*)(Bs + ((kt * 4 + t) * 32 + lane) * 16);
      v8h b0 = bp[0], b1 = bp[1];
      v16h b;
#pragma unroll
      for (int i = 0; i < 8; ++i) { b[i] = b0[i]; b[i + 8] = b1[i]; }
      acc[t] = __builtin_amdgcn_wmma_f32_16x16x32_f16(
          /*neg_a=*/false, a, /*neg_b=*/false, b,
          /*c_mod=*/(short)0, acc[t], /*reuse_a=*/false, /*reuse_b=*/false);
    }
  }

  // C/D layout: VGPR r -> (M = r + 8*half, N = lane&15)
  const int colBase = n64 * 64;
#pragma unroll
  for (int t = 0; t < 4; ++t) {
#pragma unroll
    for (int r = 0; r < 8; ++r) {
      int row = mtile * 16 + r + half * 8;
      if (row >= M) continue;
      int col = colBase + t * 16 + l16;
      float v = acc[t][r] + bias[col];
      if (MODE == EP_VALUE) {
        int b = row / NVQ, pix = row - b * NVQ;
        int h = col >> 5, ch = col & 31;
        if (mask[row]) v = 0.f;
        outF32[(((size_t)(b * HEADS + h)) * NVQ + pix) * HD + ch] = v;
      } else if (MODE == EP_OFF) {
        outF32[(size_t)row * 256 + col] = v;
      } else if (MODE == EP_ATTN) {
        outF32[(size_t)row * 128 + col] = v;
      } else if (MODE == EP_OUT) {
        // + identity residual (query BEFORE positional encoding), f16 for FFN GEMM
        outF16[(size_t)row * 256 + col] = (_Float16)(v + resid[(size_t)row * 256 + col]);
      } else {  // EP_FFN
        outF32[(size_t)row * 256 + col] = v;
      }
    }
  }
}

// ---------------- softmax over (LEVELS*POINTS)=16 per (row, head) ----------------
__global__ void softmax_kernel(float* __restrict__ buf, int total) {
  int t = blockIdx.x * 256 + threadIdx.x;
  if (t >= total) return;
  float* p = buf + (size_t)(t >> 3) * 128 + (t & 7) * 16;
  float e[16];
  float mx = p[0];
#pragma unroll
  for (int i = 1; i < 16; ++i) mx = fmaxf(mx, p[i]);
  float s = 0.f;
#pragma unroll
  for (int i = 0; i < 16; ++i) { e[i] = expf(p[i] - mx); s += e[i]; }
  float inv = 1.f / s;
#pragma unroll
  for (int i = 0; i < 16; ++i) p[i] = e[i] * inv;
}

// ---------------- deformable sampling: one wave per (b, query, head); lane = channel ----
__global__ __launch_bounds__(256)
void sample_kernel(const float* __restrict__ vproj,   // [b][head][pix][32]
                   const float* __restrict__ offb,    // [row][256]
                   const float* __restrict__ aw,      // [row][128] (post-softmax)
                   const float* __restrict__ ref,     // [row][4][2]
                   const int* __restrict__ shapes,    // [4][2] (H, W)
                   const int* __restrict__ starts,    // [4]
                   _Float16* __restrict__ samp) {     // [row][256] f16
  const int lane = threadIdx.x & 31;
  const int wid  = threadIdx.x >> 5;
  const long w = (long)blockIdx.x * 8 + wid;
  const long TOT = (long)ROWS * HEADS;
  if (w >= TOT) return;
  const int  h   = (int)(w % HEADS);
  const long row = w / HEADS;
  const int  b   = (int)(row / NVQ);
  const float* vb   = vproj + ((size_t)(b * HEADS + h)) * NVQ * HD;
  const float* offr = offb + (size_t)row * 256;
  const float* awr  = aw   + (size_t)row * 128;
  float acc = 0.f;
  for (int l = 0; l < LEVELS; ++l) {
    const int H  = shapes[l * 2 + 0];
    const int W  = shapes[l * 2 + 1];
    const int st = starts[l];
    const float rx = ref[((size_t)row * LEVELS + l) * 2 + 0];
    const float ry = ref[((size_t)row * LEVELS + l) * 2 + 1];
    const float fW = (float)W, fH = (float)H;
#pragma unroll
    for (int p = 0; p < POINTS; ++p) {
      const int oi = ((h * LEVELS + l) * POINTS + p) * 2;
      const float x = (rx + offr[oi]     / fW) * fW - 0.5f;
      const float y = (ry + offr[oi + 1] / fH) * fH - 0.5f;
      const float x0f = floorf(x), y0f = floorf(y);
      const float fx = x - x0f, fy = y - y0f;
      const int x0 = (int)x0f, y0 = (int)y0f;
      const float wgt = awr[(h * LEVELS + l) * POINTS + p];
      float s = 0.f;
#pragma unroll
      for (int cy = 0; cy < 2; ++cy) {
#pragma unroll
        for (int cx = 0; cx < 2; ++cx) {
          const int xi = x0 + cx, yi = y0 + cy;
          const bool valid = (xi >= 0) && (xi < W) && (yi >= 0) && (yi < H);
          const int xc = min(max(xi, 0), W - 1);
          const int yc = min(max(yi, 0), H - 1);
          const float val = vb[((size_t)(st + yc * W + xc)) * HD + lane]; // 128B/wave line
          const float cw = (cx ? fx : 1.f - fx) * (cy ? fy : 1.f - fy);
          s += valid ? val * cw : 0.f;
        }
      }
      acc += wgt * s;
    }
  }
  samp[(size_t)row * 256 + h * HD + lane] = (_Float16)acc;
}

// ---------------- launch ----------------
extern "C" void kernel_launch(void* const* d_in, const int* in_sizes, int n_in,
                              void* d_out, int out_size, void* d_ws, size_t ws_size,
                              hipStream_t stream) {
  const float* query = (const float*)d_in[0];
  /* d_in[1] = key: unused by reference */
  const float* value = (const float*)d_in[2];
  const float* qpos  = (const float*)d_in[3];
  const float* ref   = (const float*)d_in[4];
  const int*   shapes = (const int*)d_in[5];
  const int*   starts = (const int*)d_in[6];
  const unsigned char* mask = (const unsigned char*)d_in[7];
  const float* Wv = (const float*)d_in[8];  const float* bv = (const float*)d_in[9];
  const float* Wo = (const float*)d_in[10]; const float* bo = (const float*)d_in[11];
  const float* Wa = (const float*)d_in[12]; const float* ba = (const float*)d_in[13];
  const float* Wu = (const float*)d_in[14]; const float* bu = (const float*)d_in[15];
  const float* Wf = (const float*)d_in[16]; const float* bf = (const float*)d_in[17];

  const size_t RC = (size_t)ROWS * 256;
  char* p = (char*)d_ws;
  _Float16* q16   = (_Float16*)p; p += RC * 2;
  _Float16* v16   = (_Float16*)p; p += RC * 2;
  float*    vproj = (float*)p;    p += RC * 4;
  float*    offb  = (float*)p;    p += RC * 4;
  float*    attn  = (float*)p;    p += (size_t)ROWS * 128 * 4;
  _Float16* samp  = (_Float16*)p; p += RC * 2;
  _Float16* tmp   = (_Float16*)p; p += RC * 2;
  _Float16* fragV = (_Float16*)p; p += 65536 * 2;
  _Float16* fragO = (_Float16*)p; p += 65536 * 2;
  _Float16* fragA = (_Float16*)p; p += 32768 * 2;
  _Float16* fragU = (_Float16*)p; p += 65536 * 2;
  _Float16* fragF = (_Float16*)p;

  const int M = ROWS;
  const int Mtiles = (M + 15) / 16;
  const int gx = (Mtiles + 3) / 4;   // 4 waves / block, one 16x64 tile each

  prep_kernel<<<(int)((RC + 255) / 256), 256, 0, stream>>>(query, qpos, value, q16, v16, (long)RC);
  wprep_kernel<<<256, 256, 0, stream>>>(Wv, fragV, 256);
  wprep_kernel<<<256, 256, 0, stream>>>(Wo, fragO, 256);
  wprep_kernel<<<128, 256, 0, stream>>>(Wa, fragA, 128);
  wprep_kernel<<<256, 256, 0, stream>>>(Wu, fragU, 256);
  wprep_kernel<<<256, 256, 0, stream>>>(Wf, fragF, 256);

  gemm_wmma_kernel<EP_VALUE><<<dim3(gx, 4), 128, 0, stream>>>(v16, fragV, bv, vproj, nullptr, nullptr, mask, M, 256);
  gemm_wmma_kernel<EP_OFF  ><<<dim3(gx, 4), 128, 0, stream>>>(q16, fragO, bo, offb, nullptr, nullptr, nullptr, M, 256);
  gemm_wmma_kernel<EP_ATTN ><<<dim3(gx, 2), 128, 0, stream>>>(q16, fragA, ba, attn, nullptr, nullptr, nullptr, M, 128);

  softmax_kernel<<<(ROWS * 8 + 255) / 256, 256, 0, stream>>>(attn, ROWS * 8);
  sample_kernel<<<(int)(((long)ROWS * HEADS + 7) / 8), 256, 0, stream>>>(vproj, offb, attn, ref, shapes, starts, samp);

  gemm_wmma_kernel<EP_OUT><<<dim3(gx, 4), 128, 0, stream>>>(samp, fragU, bu, nullptr, tmp, query, nullptr, M, 256);
  gemm_wmma_kernel<EP_FFN><<<dim3(gx, 4), 128, 0, stream>>>(tmp, fragF, bf, (float*)d_out, nullptr, nullptr, nullptr, M, 256);
}
